// BI3Block_2181843386500
// MI455X (gfx1250) — matrix operand order repaired and checked by reading, not compiled, not run on hardware
//
#include <hip/hip_runtime.h>
#include <hip/hip_bf16.h>
#include <math.h>

// ---------------------------------------------------------------------------
// BI3 block on gfx1250: all GEMMs via v_wmma_f32_16x16x32_f16 (wave32 WMMA).
// Each GEMM wave computes a 16x64 strip (4 accumulators) to reuse the A
// fragment across 4 WMMA issues.
// ---------------------------------------------------------------------------

typedef __attribute__((ext_vector_type(16))) _Float16 v16h;
typedef __attribute__((ext_vector_type(8)))  float    v8f;

#define BB   8
#define NN   1024
#define CC   256
#define HWD  32
#define HEADS 8
#define DH   32
#define MROWS (BB*NN)   // 8192
#define EPSV 1e-5f

// ---- WMMA fragment loaders (layouts per CDNA5 ISA 7.12.2, wave32) ----------

// A fragment: 16x32 (MxK), A row-major with leading dim ld, base = &A[row0*ld + k0]
__device__ __forceinline__ v16h load_fragA(const _Float16* __restrict__ base, int ld) {
    int lane = threadIdx.x & 31;
    int m = lane & 15, hi = lane >> 4;
    v16h f;
#pragma unroll
    for (int v = 0; v < 8; ++v) {
        int kb = ((v & 4) << 2) + (hi << 3) + ((v & 3) << 1); // (v>=4?16:0)+(hi?8:0)+2*(v&3)
        const _Float16* p = base + (size_t)m * ld + kb;
        f[2 * v]     = p[0];
        f[2 * v + 1] = p[1];
    }
    return f;
}

// B fragment: 32x16 (KxN), element B[k][n] = W[(col0+n)*ld + k0 + k]
// (W stored row-major (O,K); base = &W[col0*ld + k0])
__device__ __forceinline__ v16h load_fragB(const _Float16* __restrict__ base, int ld) {
    int lane = threadIdx.x & 31;
    int n = lane & 15, hi = lane >> 4;
    v16h f;
#pragma unroll
    for (int v = 0; v < 8; ++v) {
        int k = ((v & 4) << 2) + (hi << 3) + ((v & 3) << 1);  // (v>=4?16:0)+(hi?8:0)+2*(v&3)
        const _Float16* p = base + (size_t)n * ld + k;
        f[2 * v]     = p[0];
        f[2 * v + 1] = p[1];
    }
    return f;
}

__device__ __forceinline__ v8f wmma_f16(v16h a, v16h b, v8f c) {
    return __builtin_amdgcn_wmma_f32_16x16x32_f16(false, a, false, b, (short)0, c, false, false);
}

__device__ __forceinline__ float gelu_exact(float x) {
    return 0.5f * x * (1.0f + erff(x * 0.70710678118654752f));
}

// ---- weight conversion f32 -> f16, optional transpose ----------------------
__global__ void cvt_w(const float* __restrict__ src, _Float16* __restrict__ dst,
                      int rows, int cols, int transpose) {
    int i = blockIdx.x * blockDim.x + threadIdx.x;
    if (i >= rows * cols) return;
    int r = i / cols, c = i % cols;
    _Float16 v = (_Float16)src[i];
    if (transpose) dst[(size_t)c * rows + r] = v;   // dst is (cols, rows)
    else           dst[i] = v;
}

// ---- LPE: depthwise 3x3 / 1x5 / 5x1 + BN + GELU -> cat (M x 768) f16 -------
__global__ void lpe_cat(const float* __restrict__ x,   // (B,N,C)
                        const float* __restrict__ w3,  const float* __restrict__ b3,
                        const float* __restrict__ w15, const float* __restrict__ b15,
                        const float* __restrict__ w51, const float* __restrict__ b51,
                        const float* __restrict__ bng, const float* __restrict__ bnb,
                        const float* __restrict__ bnm, const float* __restrict__ bnv,
                        _Float16* __restrict__ cat) {
    int row = blockIdx.x;            // b*1024 + n
    int c   = threadIdx.x;           // 0..255
    int b   = row >> 10;
    int nn  = row & 1023;
    int hh  = nn >> 5, ww = nn & 31;
    const float* xb = x + (size_t)b * NN * CC;

    auto X = [&](int h, int w) -> float {
        if (h < 0 || h > 31 || w < 0 || w > 31) return 0.f;
        return xb[(size_t)(h * HWD + w) * CC + c];
    };

    float a = b3[c];
#pragma unroll
    for (int dh = -1; dh <= 1; ++dh)
#pragma unroll
        for (int dw = -1; dw <= 1; ++dw)
            a += X(hh + dh, ww + dw) * w3[c * 9 + (dh + 1) * 3 + (dw + 1)];

    float hz = b15[c];
#pragma unroll
    for (int dw = -2; dw <= 2; ++dw) hz += X(hh, ww + dw) * w15[c * 5 + dw + 2];

    float vz = b51[c];
#pragma unroll
    for (int dh = -2; dh <= 2; ++dh) vz += X(hh + dh, ww) * w51[c * 5 + dh + 2];

    size_t base = (size_t)row * 768;
#pragma unroll
    for (int part = 0; part < 3; ++part) {
        int ch = part * CC + c;
        float t = (part == 0) ? a : (part == 1) ? hz : vz;
        float inv = bng[ch] * rsqrtf(bnv[ch] + EPSV);
        float y = (t - bnm[ch]) * inv + bnb[ch];
        cat[base + ch] = (_Float16)gelu_exact(y);
    }
}

// ---- elementwise mod = x1s * (x2s - x1s) -> f16 ----------------------------
__global__ void mod_kernel(const float* __restrict__ a, const float* __restrict__ b,
                           _Float16* __restrict__ out) {
    int i = blockIdx.x * blockDim.x + threadIdx.x;
    float q = a[i];
    out[i] = (_Float16)(q * (b[i] - q));
}

// ---- generic WMMA GEMM: Y[M,O] = A[M,K] * W[O,K]^T, fused epilogues --------
// Wave computes 16 rows x 64 cols (4 accumulators, shared A fragment).
// Block = 4 waves -> 16 x 256 outputs per block; grid = (M/16, O/256).
// MODE 0: y = acc + bias + res  -> out32 (and out16 if non-null)
// MODE 1: y = relu(BN(acc + bias)) with p0=g p1=beta p2=m p3=v -> out16
// MODE 2: y = acc + bias -> out16 in (B,H,N,32) head layout
// MODE 3: y = gelu(acc + bias) -> out16
template <int MODE>
__global__ __launch_bounds__(128)
void gemm16(const _Float16* __restrict__ A, const _Float16* __restrict__ W,
            const float* __restrict__ bias, int K,
            const float* __restrict__ res,
            const float* __restrict__ p0, const float* __restrict__ p1,
            const float* __restrict__ p2, const float* __restrict__ p3,
            float* __restrict__ out32, _Float16* __restrict__ out16, int O) {
    int wave = threadIdx.x >> 5;
    int row0 = blockIdx.x * 16;
    int col0 = (blockIdx.y * 4 + wave) * 64;

    v8f acc[4] = {{}, {}, {}, {}};
    const _Float16* Abase = A + (size_t)row0 * K;
    for (int k0 = 0; k0 < K; k0 += 32) {
        v16h af = load_fragA(Abase + k0, K);
#pragma unroll
        for (int t = 0; t < 4; ++t) {
            v16h bf = load_fragB(W + (size_t)(col0 + 16 * t) * K + k0, K);
            acc[t] = wmma_f16(af, bf, acc[t]);
        }
    }

    int lane = threadIdx.x & 31;
    int n = lane & 15, hi = lane >> 4;
#pragma unroll
    for (int t = 0; t < 4; ++t) {
#pragma unroll
        for (int r = 0; r < 8; ++r) {
            int m = r + (hi << 3);
            int row = row0 + m;
            int o   = col0 + 16 * t + n;
            float y = acc[t][r] + bias[o];
            if (MODE == 0) {
                y += res[(size_t)row * O + o];
                out32[(size_t)row * O + o] = y;
                if (out16) out16[(size_t)row * O + o] = (_Float16)y;
            } else if (MODE == 1) {
                float inv = p0[o] * rsqrtf(p3[o] + EPSV);
                y = (y - p2[o]) * inv + p1[o];
                y = fmaxf(y, 0.f);
                out16[(size_t)row * O + o] = (_Float16)y;
            } else if (MODE == 2) {
                int bb = row >> 10, nn2 = row & 1023;
                int h = o >> 5, d = o & 31;
                out16[((size_t)(bb * HEADS + h) * NN + nn2) * DH + d] = (_Float16)y;
            } else {
                out16[(size_t)row * O + o] = (_Float16)gelu_exact(y);
            }
        }
    }
}

// ---- attention: per (b,h) x 16-query tile, scores strip in LDS -------------
// Q/K/V in (B,H,N,32) f16; out -> (B,N,C) f16 with c = h*32+d
__global__ void attn_kernel(const _Float16* __restrict__ Qh,
                            const _Float16* __restrict__ Kh,
                            const _Float16* __restrict__ Vh,
                            _Float16* __restrict__ outp) {
    extern __shared__ float sm[];                 // 16*1024 scores + 16 rowsum
    float* rowsum = sm + 16 * 1024;
    int bh = blockIdx.y;
    int qt = blockIdx.x;
    const _Float16* Q  = Qh + ((size_t)bh * NN + qt * 16) * DH;
    const _Float16* Kb = Kh + (size_t)bh * NN * DH;
    const _Float16* Vb = Vh + (size_t)bh * NN * DH;

    int lane = threadIdx.x & 31;
    int n = lane & 15, hi = lane >> 4;
    const float scale = 0.17677669529663687f;     // 1/sqrt(32)

    v16h qf = load_fragA(Q, DH);                  // 16x32, whole head dim in one K step

    // S = Q K^T, streamed to LDS
    for (int kc = 0; kc < 64; ++kc) {
        v16h kf = load_fragB(Kb + (size_t)kc * 16 * DH, DH);
        v8f s = {};
        s = wmma_f16(qf, kf, s);
#pragma unroll
        for (int r = 0; r < 8; ++r) {
            int m = r + (hi << 3);
            sm[m * 1024 + kc * 16 + n] = s[r] * scale;
        }
    }
    __syncthreads();

    // softmax over 1024 keys: lane owns row (lane&15), half of the cols
    {
        int rrow = lane & 15;
        float* rp = sm + rrow * 1024 + hi * 512;
        float mx = -3.4e38f;
        for (int j = 0; j < 512; ++j) mx = fmaxf(mx, rp[j]);
        mx = fmaxf(mx, __shfl_xor(mx, 16, 32));
        float ssum = 0.f;
        for (int j = 0; j < 512; ++j) {
            float e = expf(rp[j] - mx);
            rp[j] = e;
            ssum += e;
        }
        ssum += __shfl_xor(ssum, 16, 32);
        if (hi == 0) rowsum[rrow] = ssum;
    }
    __syncthreads();

    // O = P V  (two 16x16 col tiles of d)
    v8f acc0 = {}, acc1 = {};
    for (int kc = 0; kc < 32; ++kc) {
        v16h pf;
        {
            int m = lane & 15;
            const float* pr = sm + m * 1024 + kc * 32;
#pragma unroll
            for (int v = 0; v < 8; ++v) {
                int kb = ((v & 4) << 2) + (hi << 3) + ((v & 3) << 1);
                pf[2 * v]     = (_Float16)pr[kb];
                pf[2 * v + 1] = (_Float16)pr[kb + 1];
            }
        }
        v16h vf0, vf1;
#pragma unroll
        for (int v = 0; v < 8; ++v) {
            int k = ((v & 4) << 2) + (hi << 3) + ((v & 3) << 1);
            const _Float16* vp = Vb + (size_t)(kc * 32 + k) * DH + n;
            vf0[2 * v]     = vp[0];
            vf0[2 * v + 1] = vp[DH];
            vf1[2 * v]     = vp[16];
            vf1[2 * v + 1] = vp[DH + 16];
        }
        acc0 = wmma_f16(pf, vf0, acc0);
        acc1 = wmma_f16(pf, vf1, acc1);
    }

    int b = bh >> 3, h = bh & 7;
#pragma unroll
    for (int r = 0; r < 8; ++r) {
        int m = r + (hi << 3);
        float inv = 1.0f / rowsum[m];
        size_t off = ((size_t)b * NN + qt * 16 + m) * CC + h * DH;
        outp[off + n]      = (_Float16)(acc0[r] * inv);
        outp[off + 16 + n] = (_Float16)(acc1[r] * inv);
    }
}

// ---- LayerNorm over C=256, one block per row -------------------------------
__global__ void ln_kernel(const float* __restrict__ src,
                          const float* __restrict__ g, const float* __restrict__ b,
                          float* __restrict__ out32, _Float16* __restrict__ out16) {
    __shared__ float red[CC];
    int row = blockIdx.x;
    int c = threadIdx.x;
    float v = src[(size_t)row * CC + c];

    red[c] = v; __syncthreads();
    for (int s = 128; s > 0; s >>= 1) { if (c < s) red[c] += red[c + s]; __syncthreads(); }
    float mu = red[0] * (1.0f / CC);
    __syncthreads();

    float d = v - mu;
    red[c] = d * d; __syncthreads();
    for (int s = 128; s > 0; s >>= 1) { if (c < s) red[c] += red[c + s]; __syncthreads(); }
    float var = red[0] * (1.0f / CC);

    float y = d * rsqrtf(var + EPSV) * g[c] + b[c];
    out32[(size_t)row * CC + c] = y;
    if (out16) out16[(size_t)row * CC + c] = (_Float16)y;
}

// ---------------------------------------------------------------------------

enum {
    IN_X1 = 0, IN_X2, IN_X3,
    LQ_W3, LQ_B3, LQ_W15, LQ_B15, LQ_W51, LQ_B51,
    LQ_BNG, LQ_BNB, LQ_BNM, LQ_BNV, LQ_FW, LQ_FB,
    LK_W3, LK_B3, LK_W15, LK_B15, LK_W51, LK_B51,
    LK_BNG, LK_BNB, LK_BNM, LK_BNV, LK_FW, LK_FB,
    G_WQ, G_BQ, G_WK, G_BK, G_WV, G_BV,
    G_DMW, G_DMB, G_DMG, G_DMBETA, G_DMM, G_DMV, G_WP, G_BP,
    F_W1, F_B1, F_W2, F_B2,
    LN1G, LN1B, LN2G, LN2B
};

extern "C" void kernel_launch(void* const* d_in, const int* in_sizes, int n_in,
                              void* d_out, int out_size, void* d_ws, size_t ws_size,
                              hipStream_t stream) {
    (void)in_sizes; (void)n_in; (void)out_size; (void)ws_size;
    auto f32 = [&](int i) { return (const float*)d_in[i]; };

    // ---- workspace carve-out (256B aligned) ----
    char* wp_ = (char*)d_ws;
    size_t off = 0;
    auto alloc = [&](size_t bytes) {
        void* p = wp_ + off;
        off += (bytes + 255) & ~(size_t)255;
        return p;
    };
    _Float16* fwq = (_Float16*)alloc((size_t)CC * 768 * 2);
    _Float16* fwk = (_Float16*)alloc((size_t)CC * 768 * 2);
    _Float16* dmw = (_Float16*)alloc((size_t)CC * CC * 2);
    _Float16* wqT = (_Float16*)alloc((size_t)CC * CC * 2);
    _Float16* wkT = (_Float16*)alloc((size_t)CC * CC * 2);
    _Float16* wvT = (_Float16*)alloc((size_t)CC * CC * 2);
    _Float16* wpT = (_Float16*)alloc((size_t)CC * CC * 2);
    _Float16* w1T = (_Float16*)alloc((size_t)1024 * CC * 2);
    _Float16* w2T = (_Float16*)alloc((size_t)CC * 1024 * 2);
    _Float16* big16  = (_Float16*)alloc((size_t)MROWS * 1024 * 2); // CAT(768) then h1(1024)
    float*    x1s32  = (float*)alloc((size_t)MROWS * CC * 4);
    float*    x2s32  = (float*)alloc((size_t)MROWS * CC * 4);      // later: xattn f32
    _Float16* x1s16  = (_Float16*)alloc((size_t)MROWS * CC * 2);
    _Float16* abuf16 = (_Float16*)alloc((size_t)MROWS * CC * 2);   // mod -> attnout -> xattn16
    _Float16* kv16   = (_Float16*)alloc((size_t)MROWS * CC * 2);
    _Float16* Qh     = (_Float16*)alloc((size_t)MROWS * CC * 2);
    _Float16* Kh     = (_Float16*)alloc((size_t)MROWS * CC * 2);
    _Float16* Vh     = (_Float16*)alloc((size_t)MROWS * CC * 2);
    float*    pre32  = (float*)alloc((size_t)MROWS * CC * 4);      // preLN1 then preLN2

    auto cvt = [&](int idx, _Float16* dst, int rows, int cols, int transpose) {
        int numel = rows * cols;
        cvt_w<<<(numel + 255) / 256, 256, 0, stream>>>(f32(idx), dst, rows, cols, transpose);
    };
    // fw / dm_w already (O,K); linear weights stored (K,O) -> transpose
    cvt(LQ_FW, fwq, CC, 768, 0);
    cvt(LK_FW, fwk, CC, 768, 0);
    cvt(G_DMW, dmw, CC, CC, 0);
    cvt(G_WQ, wqT, CC, CC, 1);
    cvt(G_WK, wkT, CC, CC, 1);
    cvt(G_WV, wvT, CC, CC, 1);
    cvt(G_WP, wpT, CC, CC, 1);
    cvt(F_W1, w1T, CC, 1024, 1);   // -> (1024, 256)
    cvt(F_W2, w2T, 1024, CC, 1);   // -> (256, 1024)

    dim3 gemmBlk(128);
    dim3 g256(MROWS / 16, 1);      // O = 256: one block covers all 256 cols
    dim3 g1024(MROWS / 16, 4);     // O = 1024

    // ---- x1s = x1 + LPE_q(x1) ----
    lpe_cat<<<MROWS, CC, 0, stream>>>(f32(IN_X1),
        f32(LQ_W3), f32(LQ_B3), f32(LQ_W15), f32(LQ_B15), f32(LQ_W51), f32(LQ_B51),
        f32(LQ_BNG), f32(LQ_BNB), f32(LQ_BNM), f32(LQ_BNV), big16);
    gemm16<0><<<g256, gemmBlk, 0, stream>>>(big16, fwq, f32(LQ_FB), 768,
        f32(IN_X1), nullptr, nullptr, nullptr, nullptr, x1s32, x1s16, CC);

    // ---- x2s = x2 + LPE_k(x2) ----
    lpe_cat<<<MROWS, CC, 0, stream>>>(f32(IN_X2),
        f32(LK_W3), f32(LK_B3), f32(LK_W15), f32(LK_B15), f32(LK_W51), f32(LK_B51),
        f32(LK_BNG), f32(LK_BNB), f32(LK_BNM), f32(LK_BNV), big16);
    gemm16<0><<<g256, gemmBlk, 0, stream>>>(big16, fwk, f32(LK_FB), 768,
        f32(IN_X2), nullptr, nullptr, nullptr, nullptr, x2s32, nullptr, CC);

    // ---- mod = x1s*(x2s - x1s); kv = relu(BN(mod @ dm_w + dm_b)) ----
    mod_kernel<<<MROWS, CC, 0, stream>>>(x1s32, x2s32, abuf16);
    gemm16<1><<<g256, gemmBlk, 0, stream>>>(abuf16, dmw, f32(G_DMB), CC,
        nullptr, f32(G_DMG), f32(G_DMBETA), f32(G_DMM), f32(G_DMV), nullptr, kv16, CC);

    // ---- q/k/v projections into (B,H,N,32) ----
    gemm16<2><<<g256, gemmBlk, 0, stream>>>(x1s16, wqT, f32(G_BQ), CC,
        nullptr, nullptr, nullptr, nullptr, nullptr, nullptr, Qh, CC);
    gemm16<2><<<g256, gemmBlk, 0, stream>>>(kv16, wkT, f32(G_BK), CC,
        nullptr, nullptr, nullptr, nullptr, nullptr, nullptr, Kh, CC);
    gemm16<2><<<g256, gemmBlk, 0, stream>>>(kv16, wvT, f32(G_BV), CC,
        nullptr, nullptr, nullptr, nullptr, nullptr, nullptr, Vh, CC);

    // ---- attention ----
    size_t lds = (16 * 1024 + 16) * sizeof(float);
    attn_kernel<<<dim3(NN / 16, BB * HEADS), 32, lds, stream>>>(Qh, Kh, Vh, abuf16);

    // ---- out proj + residual, LN1 ----
    gemm16<0><<<g256, gemmBlk, 0, stream>>>(abuf16, wpT, f32(G_BP), CC,
        x1s32, nullptr, nullptr, nullptr, nullptr, pre32, nullptr, CC);
    ln_kernel<<<MROWS, CC, 0, stream>>>(pre32, f32(LN1G), f32(LN1B), x2s32, abuf16);

    // ---- FFN ----
    gemm16<3><<<g1024, gemmBlk, 0, stream>>>(abuf16, w1T, f32(F_B1), CC,
        nullptr, nullptr, nullptr, nullptr, nullptr, nullptr, big16, 1024);
    gemm16<0><<<g256, gemmBlk, 0, stream>>>(big16, w2T, f32(F_B2), 1024,
        x2s32, nullptr, nullptr, nullptr, nullptr, pre32, nullptr, CC);

    // ---- LN2 -> output (f32) ----
    ln_kernel<<<MROWS, CC, 0, stream>>>(pre32, f32(LN2G), f32(LN2B), (float*)d_out, nullptr);
}
